// GruAttentionCritic_75350906241221
// MI455X (gfx1250) — matrix operand order repaired and checked by reading, not compiled, not run on hardware
//
#include <hip/hip_runtime.h>
#include <hip/hip_bf16.h>

// Problem constants
constexpr int NT    = 16;      // time steps
constexpr int BT    = 32768;   // batch
constexpr int SDIM  = 128;     // state dim
constexpr int ADm   = 8;       // action dim
constexpr int HD    = 128;     // hidden
constexpr int TB    = 64;      // batch rows per workgroup
constexpr int KA    = 160;     // padded K for [states|actions|pad] GEMM (5 * 32)

typedef __attribute__((ext_vector_type(16))) __bf16         v16bf;
typedef __attribute__((ext_vector_type(8)))  float          v8f;
typedef __attribute__((ext_vector_type(8)))  unsigned short ushort8;
typedef __attribute__((ext_vector_type(4)))  unsigned int   u32x4;
typedef __attribute__((ext_vector_type(8)))  int            i32x8;
typedef __attribute__((ext_vector_type(4)))  int            i32x4;

union Frag { v16bf v; ushort8 u[2]; };

// -------- LDS pool layout (phase 1: encoder/attention loop) --------
constexpr int OFF_WENC = 0;                        // 128x160 bf16 = 40960
constexpr int OFF_WK   = 40960;                    // 128x128 bf16 = 32768
constexpr int OFF_WV   = 73728;                    // 128x128 bf16 = 32768
constexpr int OFF_A    = 106496;                   // 64x160 bf16  = 20480 (aliased as E 64x128)
constexpr int OFF_K    = 126976;                   // 64x128 bf16  = 16384
constexpr int OFF_V    = 143360;                   // 64x128 bf16  = 16384
constexpr int OFF_ACT  = 159744;                   // 64x8 f32     = 2048  (TDM target)
constexpr int POOL_BYTES = 161792;                 // ~158 KB
// -------- phase 2 (post-loop) aliases --------
constexpr int OFF_A2   = 0;                        // 64x256 bf16 = 32768
constexpr int OFF_WC   = 32768;                    // 128x256 bf16 = 65536 (also gate slices 128x128)
constexpr int OFF_H1   = 98304;                    // 64x128 bf16 = 16384
constexpr int OFF_H0   = 114688;                   // 64x128 bf16 = 16384

__device__ __forceinline__ unsigned short f2bf(float x) {
  unsigned u = __builtin_bit_cast(unsigned, x);
  u = (u + 0x7FFFu + ((u >> 16) & 1u)) >> 16;
  return (unsigned short)u;
}
__device__ __forceinline__ float bf2f(unsigned short h) {
  return __builtin_bit_cast(float, ((unsigned)h) << 16);
}
__device__ __forceinline__ float lrelu(float x) { return x > 0.f ? x : 0.01f * x; }
__device__ __forceinline__ float sigm(float x)  { return 1.f / (1.f + __expf(-x)); }

__device__ __forceinline__ void zero4(v8f a[4]) {
#pragma unroll
  for (int t = 0; t < 4; ++t) a[t] = (v8f){0.f,0.f,0.f,0.f,0.f,0.f,0.f,0.f};
}

// D = A(64xK, LDS bf16 row-major) @ W^T where W is (N=128 x K) LDS bf16 row-major.
// Each wave computes rows [rowBase,rowBase+16) x cols [colBase,colBase+64).
template<int KSTEPS>
__device__ __forceinline__ void gemm_bf16(const unsigned short* A, int strideA,
                                          const unsigned short* W, int strideW,
                                          int rowBase, int colBase, int lane,
                                          v8f acc[4]) {
  const int m = lane & 15, hf = lane >> 4;
#pragma unroll
  for (int ks = 0; ks < KSTEPS; ++ks) {
    Frag a;
    const unsigned short* ap = A + (rowBase + m) * strideA + ks * 32 + 8 * hf;
    a.u[0] = *(const ushort8*)(ap);        // K = 32ks + 8h .. +7
    a.u[1] = *(const ushort8*)(ap + 16);   // K = 32ks + 16 + 8h .. +7
#pragma unroll
    for (int t = 0; t < 4; ++t) {
      Frag b;
      const unsigned short* wp = W + (colBase + t * 16 + m) * strideW + ks * 32 + 8 * hf;
      b.u[0] = *(const ushort8*)(wp);      // K = 32ks + 8h .. +7
      b.u[1] = *(const ushort8*)(wp + 16); // K = 32ks + 16 + 8h .. +7
      acc[t] = __builtin_amdgcn_wmma_f32_16x16x32_bf16(false, a.v, false, b.v,
                                                       (short)0, acc[t], false, false);
    }
  }
}

// Issue a 2-D TDM load of f32 data: tile (tile0 x tile1) from a (td0-wide, row
// stride stride0) tensor into LDS at byte offset lds_off. D# per ISA 08.
// clang-23 therock form: (u32x4, i32x8, i32x4, i32x4, i32x8, i32 cpol).
__device__ __forceinline__ void tdm_load_2d(unsigned lds_off, const float* gptr,
                                            unsigned tile0, unsigned tile1,
                                            unsigned td0, unsigned td1,
                                            unsigned long long stride0) {
  unsigned long long ga = (unsigned long long)(const void*)gptr;
  u32x4 g0;
  g0[0] = 1u;                                               // count = 1 (valid)
  g0[1] = lds_off;                                          // lds_addr
  g0[2] = (unsigned)(ga & 0xFFFFFFFFull);                   // global_addr[31:0]
  g0[3] = (unsigned)((ga >> 32) & 0x1FFFFFFull) | (2u << 30); // [56:32] | type=2
  i32x8 g1;
  g1[0] = (int)(2u << 16);                                  // data_size = 4B
  g1[1] = (int)((td0 & 0xFFFFu) << 16);                     // tensor_dim0 lo
  g1[2] = (int)((td0 >> 16) | ((td1 & 0xFFFFu) << 16));     // dim0 hi | dim1 lo
  g1[3] = (int)((td1 >> 16) | ((tile0 & 0xFFFFu) << 16));   // dim1 hi | tile_dim0
  g1[4] = (int)(tile1 & 0xFFFFu);                           // tile_dim1 (tile_dim2=0)
  g1[5] = (int)(stride0 & 0xFFFFFFFFull);                   // tensor_dim0_stride lo
  g1[6] = (int)((stride0 >> 32) & 0xFFFFull);               // stride hi (dim1_stride=0)
  g1[7] = 0;
  i32x4 gz4 = {0, 0, 0, 0};
  i32x8 gz8 = {0, 0, 0, 0, 0, 0, 0, 0};
  __builtin_amdgcn_tensor_load_to_lds(g0, g1, gz4, gz4, gz8, 0);
}

__global__ __launch_bounds__(256, 1)
void gac_fused_kernel(const float* __restrict__ states, const float* __restrict__ actions,
                      const float* __restrict__ h0,     const float* __restrict__ Wenc,
                      const float* __restrict__ benc,   const float* __restrict__ Ws,
                      const float* __restrict__ bs,     const float* __restrict__ Wk,
                      const float* __restrict__ Wsel,   const float* __restrict__ Wv,
                      const float* __restrict__ bv,     const float* __restrict__ Wc,
                      const float* __restrict__ bc,     const float* __restrict__ Wih,
                      const float* __restrict__ Whh,    const float* __restrict__ bih,
                      const float* __restrict__ bhh,    const float* __restrict__ Wfc,
                      const float* __restrict__ bfc,    float* __restrict__ out,
                      unsigned short* __restrict__ ws_senc) {
  __shared__ __align__(16) unsigned char pool[POOL_BYTES];
  const int tid  = threadIdx.x;
  const int lane = tid & 31, wave = tid >> 5;
  const int rowBase = (wave & 3) * 16;
  const int colBase = (wave >> 2) * 64;
  const int b0 = blockIdx.x * TB;
  const int m = lane & 15, hf = lane >> 4;
  const unsigned lds_act = (unsigned)(unsigned long long)(const void*)(pool + OFF_ACT);

  unsigned short* sWENC = (unsigned short*)(pool + OFF_WENC);
  unsigned short* sWK   = (unsigned short*)(pool + OFF_WK);
  unsigned short* sWV   = (unsigned short*)(pool + OFF_WV);
  unsigned short* sA    = (unsigned short*)(pool + OFF_A);
  unsigned short* sK    = (unsigned short*)(pool + OFF_K);
  unsigned short* sV    = (unsigned short*)(pool + OFF_V);
  float*          sAct  = (float*)(pool + OFF_ACT);

  // ---------------- preamble: s_enc = lrelu(states[0] @ Ws^T + bs) ----------------
  for (int i = tid; i < HD * HD; i += 256) sWK[i] = f2bf(Ws[i]);   // Ws into WK slot
  for (int i = tid; i < TB * SDIM; i += 256) {
    int r = i >> 7, c = i & 127;
    sA[r * KA + c] = f2bf(states[(size_t)(b0 + r) * SDIM + c]);    // states[0] tile
  }
  __syncthreads();
  {
    v8f acc[4]; zero4(acc);
    gemm_bf16<4>(sA, KA, sWK, HD, rowBase, colBase, lane, acc);
#pragma unroll
    for (int t = 0; t < 4; ++t)
#pragma unroll
      for (int j = 0; j < 8; ++j) {
        int row = rowBase + j + 8 * hf, col = colBase + t * 16 + m;
        unsigned short b = f2bf(lrelu(acc[t][j] + bs[col]));
        sK[row * HD + col] = b;                                    // s_enc bf16 (temp in K slot)
        ws_senc[(size_t)(b0 + row) * HD + col] = b;                // park in global scratch
      }
  }
  __syncthreads();
  // ---------------- sel = s_enc @ Wsel^T ----------------
  for (int i = tid; i < HD * HD; i += 256) sWV[i] = f2bf(Wsel[i]); // Wsel into WV slot
  __syncthreads();
  {
    v8f acc[4]; zero4(acc);
    gemm_bf16<4>(sK, HD, sWV, HD, rowBase, colBase, lane, acc);
#pragma unroll
    for (int t = 0; t < 4; ++t)
#pragma unroll
      for (int j = 0; j < 8; ++j) {
        int row = rowBase + j + 8 * hf, col = colBase + t * 16 + m;
        sV[row * HD + col] = f2bf(acc[t][j]);
      }
  }
  __syncthreads();
  // Each thread owns one (b, head) pair for attention.
  const int ab = tid >> 2, ae = tid & 3;
  float selr[32];
#pragma unroll
  for (int a = 0; a < 32; ++a) selr[a] = bf2f(sV[ab * HD + ae * 32 + a]);

  // ---------------- persistent weights for the n-loop ----------------
  for (int i = tid; i < HD * KA; i += 256) {                       // W_enc padded 136->160
    int r = i / KA, c = i - r * KA;
    sWENC[i] = (c < SDIM + ADm) ? f2bf(Wenc[r * (SDIM + ADm) + c]) : (unsigned short)0;
  }
  for (int i = tid; i < HD * HD; i += 256) sWK[i] = f2bf(Wk[i]);
  for (int i = tid; i < HD * HD; i += 256) sWV[i] = f2bf(Wv[i]);
  __syncthreads();

  // ---------------- attention over n = 1..15 (online softmax) ----------------
  float accA[32];
#pragma unroll
  for (int a = 0; a < 32; ++a) accA[a] = 0.f;
  float mrun = -1e30f, lrun = 0.f;

  for (int n = 1; n < NT; ++n) {
    if (wave == 0) {                                               // TDM: actions[n] tile -> LDS
      tdm_load_2d(lds_act, actions + ((size_t)n * BT + b0) * ADm,
                  ADm, TB, ADm, (unsigned)BT, (unsigned long long)ADm);
      __builtin_amdgcn_s_wait_tensorcnt(0);
    }
    const float* sp = states + ((size_t)n * BT + b0) * SDIM;
    for (int i = tid; i < TB * SDIM; i += 256) {
      int r = i >> 7, c = i & 127;
      sA[r * KA + c] = f2bf(sp[(size_t)r * SDIM + c]);
    }
    for (int i = tid; i < TB * 24; i += 256) {                     // zero pad cols 136..159
      int r = i / 24, c = i - r * 24;
      sA[r * KA + 136 + c] = 0;
    }
    if (n + 1 < NT)                                                // prefetch next states tile
      __builtin_prefetch(states + ((size_t)(n + 1) * BT + b0) * SDIM + tid * 32, 0, 0);
    __syncthreads();                                               // TDM done + states staged
    for (int i = tid; i < TB * ADm; i += 256) {
      int r = i >> 3, c = i & 7;
      sA[r * KA + SDIM + c] = f2bf(sAct[i]);
    }
    __syncthreads();

    // enc = lrelu([states|actions] @ Wenc^T + benc)
    v8f eacc[4]; zero4(eacc);
    gemm_bf16<5>(sA, KA, sWENC, KA, rowBase, colBase, lane, eacc);
    __syncthreads();                                               // done reading A; reuse as E
    unsigned short* sE = sA;                                       // 64x128 stride HD
#pragma unroll
    for (int t = 0; t < 4; ++t)
#pragma unroll
      for (int j = 0; j < 8; ++j) {
        int row = rowBase + j + 8 * hf, col = colBase + t * 16 + m;
        sE[row * HD + col] = f2bf(lrelu(eacc[t][j] + benc[col]));
      }
    __syncthreads();

    // keys = enc @ Wk^T ; values = lrelu(enc @ Wv^T + bv)
    {
      v8f kacc[4]; zero4(kacc);
      gemm_bf16<4>(sE, HD, sWK, HD, rowBase, colBase, lane, kacc);
#pragma unroll
      for (int t = 0; t < 4; ++t)
#pragma unroll
        for (int j = 0; j < 8; ++j) {
          int row = rowBase + j + 8 * hf, col = colBase + t * 16 + m;
          sK[row * HD + col] = f2bf(kacc[t][j]);
        }
      v8f vacc[4]; zero4(vacc);
      gemm_bf16<4>(sE, HD, sWV, HD, rowBase, colBase, lane, vacc);
#pragma unroll
      for (int t = 0; t < 4; ++t)
#pragma unroll
        for (int j = 0; j < 8; ++j) {
          int row = rowBase + j + 8 * hf, col = colBase + t * 16 + m;
          sV[row * HD + col] = f2bf(lrelu(vacc[t][j] + bv[col]));
        }
    }
    __syncthreads();

    // online softmax update, one (b,e) per thread
    {
      const unsigned short* kr = sK + ab * HD + ae * 32;
      const unsigned short* vr = sV + ab * HD + ae * 32;
      float lg = 0.f;
#pragma unroll
      for (int a = 0; a < 32; ++a) lg += selr[a] * bf2f(kr[a]);
      lg *= 0.17677669529663687f;                                  // 1/sqrt(32)
      float mn = fmaxf(mrun, lg);
      float al = __expf(mrun - mn);
      float p  = __expf(lg - mn);
      lrun = lrun * al + p;
#pragma unroll
      for (int a = 0; a < 32; ++a) accA[a] = accA[a] * al + p * bf2f(vr[a]);
      mrun = mn;
    }
    __syncthreads();                                               // before next-iter overwrite
  }

  // ---------------- h1 = [s_enc | other] @ Wc^T + bc ----------------
  unsigned short* pA2 = (unsigned short*)(pool + OFF_A2);
  unsigned short* pWC = (unsigned short*)(pool + OFF_WC);
  unsigned short* pWG = (unsigned short*)(pool + OFF_WC);
  unsigned short* pH1 = (unsigned short*)(pool + OFF_H1);
  unsigned short* pH0 = (unsigned short*)(pool + OFF_H0);

  for (int i = tid; i < TB * HD; i += 256) {                       // s_enc back from scratch
    int r = i >> 7, c = i & 127;
    pA2[r * 256 + c] = ws_senc[(size_t)(b0 + r) * HD + c];
  }
  {
    float inv = 1.f / lrun;
#pragma unroll
    for (int a = 0; a < 32; ++a)
      pA2[ab * 256 + 128 + ae * 32 + a] = f2bf(accA[a] * inv);     // normalized attention out
  }
  for (int i = tid; i < HD * 2 * HD; i += 256) pWC[i] = f2bf(Wc[i]);
  for (int i = tid; i < TB * HD; i += 256) {
    int r = i >> 7, c = i & 127;
    pH0[i] = f2bf(h0[(size_t)(b0 + r) * HD + c]);
  }
  __syncthreads();
  {
    v8f acc[4]; zero4(acc);
    gemm_bf16<8>(pA2, 256, pWC, 256, rowBase, colBase, lane, acc);
#pragma unroll
    for (int t = 0; t < 4; ++t)
#pragma unroll
      for (int j = 0; j < 8; ++j) {
        int row = rowBase + j + 8 * hf, col = colBase + t * 16 + m;
        pH1[row * HD + col] = f2bf(acc[t][j] + bc[col]);
      }
  }
  __syncthreads();

  // ---------------- GRU gates (r,z live in WMMA register layout) ----------------
  v8f rg[4], zg[4];
#pragma unroll 1
  for (int g = 0; g < 2; ++g) {                                    // g=0: r, g=1: z
    for (int i = tid; i < HD * HD; i += 256) pWG[i] = f2bf(Wih[(size_t)(g * HD) * HD + i]);
    __syncthreads();
    v8f acc[4]; zero4(acc);
    gemm_bf16<4>(pH1, HD, pWG, HD, rowBase, colBase, lane, acc);
    __syncthreads();
    for (int i = tid; i < HD * HD; i += 256) pWG[i] = f2bf(Whh[(size_t)(g * HD) * HD + i]);
    __syncthreads();
    gemm_bf16<4>(pH0, HD, pWG, HD, rowBase, colBase, lane, acc);   // accumulate gi+gh
#pragma unroll
    for (int t = 0; t < 4; ++t)
#pragma unroll
      for (int j = 0; j < 8; ++j) {
        int col = colBase + t * 16 + m;
        float v = sigm(acc[t][j] + bih[g * HD + col] + bhh[g * HD + col]);
        if (g == 0) rg[t][j] = v; else zg[t][j] = v;
      }
    __syncthreads();
  }
  {                                                                // n gate + combine
    for (int i = tid; i < HD * HD; i += 256) pWG[i] = f2bf(Wih[(size_t)(2 * HD) * HD + i]);
    __syncthreads();
    v8f gi[4]; zero4(gi);
    gemm_bf16<4>(pH1, HD, pWG, HD, rowBase, colBase, lane, gi);
    __syncthreads();
    for (int i = tid; i < HD * HD; i += 256) pWG[i] = f2bf(Whh[(size_t)(2 * HD) * HD + i]);
    __syncthreads();
    v8f gh[4]; zero4(gh);
    gemm_bf16<4>(pH0, HD, pWG, HD, rowBase, colBase, lane, gh);
#pragma unroll
    for (int t = 0; t < 4; ++t)
#pragma unroll
      for (int j = 0; j < 8; ++j) {
        int row = rowBase + j + 8 * hf, col = colBase + t * 16 + m;
        float gin = gi[t][j] + bih[2 * HD + col];
        float ghn = gh[t][j] + bhh[2 * HD + col];
        float nv  = tanhf(gin + rg[t][j] * ghn);
        float zv  = zg[t][j];
        float h0v = h0[(size_t)(b0 + row) * HD + col];
        pH1[row * HD + col] = f2bf((1.f - zv) * nv + zv * h0v);    // h (reuse H1 slot)
      }
  }
  // ---------------- q = lrelu(h @ Wfc^T + bfc)[argmax(actions[0])] ----------------
  if (wave == 0) {                                                 // TDM: actions[0] tile
    tdm_load_2d(lds_act, actions + (size_t)b0 * ADm,
                ADm, TB, ADm, (unsigned)BT, (unsigned long long)ADm);
    __builtin_amdgcn_s_wait_tensorcnt(0);
  }
  __syncthreads();
  if (tid < TB) {
    const float* ar = sAct + tid * ADm;
    int best = 0; float bvv = ar[0];
#pragma unroll
    for (int j = 1; j < ADm; ++j) { float v = ar[j]; if (v > bvv) { bvv = v; best = j; } }
    const unsigned short* hr = pH1 + tid * HD;
    float dot = 0.f;
    for (int c = 0; c < HD; ++c) dot += bf2f(hr[c]) * Wfc[best * HD + c];
    out[b0 + tid] = lrelu(dot + bfc[best]);
  }
}

extern "C" void kernel_launch(void* const* d_in, const int* in_sizes, int n_in,
                              void* d_out, int out_size, void* d_ws, size_t ws_size,
                              hipStream_t stream) {
  (void)in_sizes; (void)n_in; (void)out_size; (void)ws_size;
  dim3 grid(BT / TB), block(256);
  gac_fused_kernel<<<grid, block, 0, stream>>>(
      (const float*)d_in[0],  (const float*)d_in[1],  (const float*)d_in[2],
      (const float*)d_in[3],  (const float*)d_in[4],  (const float*)d_in[5],
      (const float*)d_in[6],  (const float*)d_in[7],  (const float*)d_in[8],
      (const float*)d_in[9],  (const float*)d_in[10], (const float*)d_in[11],
      (const float*)d_in[12], (const float*)d_in[13], (const float*)d_in[14],
      (const float*)d_in[15], (const float*)d_in[16], (const float*)d_in[17],
      (const float*)d_in[18], (float*)d_out, (unsigned short*)d_ws);
}
